// LSTM_Generator_25520695673371
// MI455X (gfx1250) — matrix operand order repaired and checked by reading, not compiled, not run on hardware
//
#include <hip/hip_runtime.h>

// ---------------- constants ----------------
#define HID 100          // hidden size
#define G4  400          // 4*H gate width
#define BSZ 512
#define TSZ 512
#define KS  264          // weight column K-stride (f16): 128 (h-seg) + 128 (x-seg) + 8 pad
#define AS  136          // LDS activation row stride (f16): 128 padded K + 8 pad
#define GS  404          // LDS gates row stride (f32)
#define CS  104          // LDS cell-state row stride (f32)
#define MT  16           // batch-tile rows per workgroup (WMMA M)
#define NWAVE 5
#define NTHR (NWAVE * 32)
#define NTPW 5           // N-tiles (of 16 cols) per wave: 5 waves * 5 = 25 tiles = 400 cols

typedef __attribute__((ext_vector_type(16))) _Float16 v16h;
typedef __attribute__((ext_vector_type(8)))  _Float16 v8h;
typedef __attribute__((ext_vector_type(8)))  float    v8f;

// ---------------- dynamic LDS layout (byte offsets) ----------------
#define OFF_HA    0              // _Float16 hA[3][16][AS]      = 13056 B
#define OFF_C     13056          // float    c [3][16][CS]      = 19968 B
#define OFF_G     33024          // float    gates[16][GS]      = 25856 B
#define OFF_BIAS  58880          // float    bias[3][400]       =  4800 B
#define OFF_WIH0  63680          // float    wih0[400]          =  1600 B
#define OFF_WP    65280          // float    wp[112] ([100]=bp) =   448 B
#define OFF_ZROW  65728          // float    zrow[16]           =    64 B
#define OFF_RACC  65792          // float    racc[16]           =    64 B
#define SMEM_BYTES 65856

// Convert f32 weights into WMMA-B-friendly f16 images:
//   W[l][n][k], n = gate column (0..399), column-major K (contiguous per column).
//   k in [0,100)        -> Whh_l[n][k]         (h-segment, padded to 128 with zeros)
//   k in [128,228), l>0 -> Wih_l[n][k-128]     (x-segment, padded to 256 with zeros)
__global__ __launch_bounds__(256) void prep_weights(
    const float* __restrict__ Whh0, const float* __restrict__ Wih1,
    const float* __restrict__ Whh1, const float* __restrict__ Wih2,
    const float* __restrict__ Whh2, _Float16* __restrict__ W) {
  const int total = 3 * G4 * KS;
  for (int idx = blockIdx.x * blockDim.x + threadIdx.x; idx < total;
       idx += gridDim.x * blockDim.x) {
    int l = idx / (G4 * KS);
    int r = idx - l * (G4 * KS);
    int g = r / KS;
    int k = r - g * KS;
    const float* Whh = (l == 0) ? Whh0 : (l == 1) ? Whh1 : Whh2;
    float v = 0.f;
    if (k < HID) {
      v = Whh[g * HID + k];
    } else if (l > 0 && k >= 128 && k < 128 + HID) {
      const float* Wih = (l == 1) ? Wih1 : Wih2;
      v = Wih[g * HID + (k - 128)];
    }
    W[idx] = (_Float16)v;
  }
}

__device__ __forceinline__ float sigmoid_fast(float x) {
  return 1.0f / (1.0f + __expf(-x));
}

// One persistent workgroup per 16-row batch tile; runs all 3 LSTM layers + the
// projection for all 512 timesteps. h/c state lives in LDS; weights stream from
// L2 as WMMA B-fragments.
__global__ __launch_bounds__(NTHR) void lstm3_fused(
    const float* __restrict__ z,          // [B][T]
    const _Float16* __restrict__ W,       // [3][400][KS] f16 (prepped)
    const float* __restrict__ b0, const float* __restrict__ b1,
    const float* __restrict__ b2,
    const float* __restrict__ wih0,       // [400] (layer0 input weight, Din=1)
    const float* __restrict__ wp,         // [100]
    const float* __restrict__ bp,         // [1]
    float* __restrict__ out)              // [B][T]
{
  extern __shared__ char smem[];
  _Float16* hA   = (_Float16*)(smem + OFF_HA);
  float*    cSt  = (float*)(smem + OFF_C);
  float*    gates= (float*)(smem + OFF_G);
  float*    biasL= (float*)(smem + OFF_BIAS);
  float*    wih0L= (float*)(smem + OFF_WIH0);
  float*    wpL  = (float*)(smem + OFF_WP);
  float*    zrow = (float*)(smem + OFF_ZROW);
  float*    racc = (float*)(smem + OFF_RACC);

  const int tid   = threadIdx.x;
  const int b0row = blockIdx.x * MT;

  // ---- init: zero states, stage biases/small weights into LDS ----
  for (int i = tid; i < 3 * MT * AS; i += NTHR) hA[i] = (_Float16)0.f;
  for (int i = tid; i < 3 * MT * CS; i += NTHR) cSt[i] = 0.f;
  for (int i = tid; i < G4; i += NTHR) {
    biasL[i]          = b0[i];
    biasL[G4 + i]     = b1[i];
    biasL[2 * G4 + i] = b2[i];
    wih0L[i]          = wih0[i];
  }
  for (int i = tid; i < 112; i += NTHR)
    wpL[i] = (i < HID) ? wp[i] : ((i == HID) ? bp[0] : 0.f);
  if (tid < MT) racc[tid] = 0.f;
  __syncthreads();

  const int wv   = tid >> 5;      // wave id 0..4
  const int lane = tid & 31;
  const int half = lane >> 4;     // 0: lanes 0-15, 1: lanes 16-31
  const int mr   = lane & 15;

  for (int t = 0; t < TSZ; ++t) {
    if (tid < MT) zrow[tid] = z[(size_t)(b0row + tid) * TSZ + t];
    __syncthreads();

    for (int l = 0; l < 3; ++l) {
      // ================= WMMA phase: gates = bias (+z*wih0) + [h|x] @ W =====
      float zr[8];
      if (l == 0) {
#pragma unroll
        for (int r = 0; r < 8; ++r) zr[r] = zrow[r + 8 * half];
      }
      v8f acc[NTPW];
#pragma unroll
      for (int j = 0; j < NTPW; ++j) {
        const int col = (wv * NTPW + j) * 16 + mr;
        const float bb = biasL[l * G4 + col];
        if (l == 0) {
          const float w0 = wih0L[col];
#pragma unroll
          for (int r = 0; r < 8; ++r) acc[j][r] = bb + w0 * zr[r];
        } else {
#pragma unroll
          for (int r = 0; r < 8; ++r) acc[j][r] = bb;
        }
      }
      const int nsteps = (l == 0) ? 4 : 8;          // h-seg only for layer 0
      const _Float16* WL = W + (size_t)l * G4 * KS;
      for (int s = 0; s < nsteps; ++s) {
        // A operand: s<4 -> h_{t-1} of this layer; s>=4 -> h_t of layer l-1.
        const _Float16* Ab = hA + ((s < 4) ? l : (l - 1)) * MT * AS;
        const int k0 = (s & 3) * 32;
        // 16-bit A 16x32 layout: lanes 0-15 hold K {k0..k0+7, k0+16..k0+23},
        // lanes 16-31 hold K {k0+8.., k0+24..} for row M = lane%16.
        v8h alo = *(const v8h*)(Ab + mr * AS + k0 + half * 8);
        v8h ahi = *(const v8h*)(Ab + mr * AS + k0 + 16 + half * 8);
        v16h a = __builtin_shufflevector(alo, ahi, 0, 1, 2, 3, 4, 5, 6, 7, 8,
                                         9, 10, 11, 12, 13, 14, 15);
        const int kb = s * 32 + half * 8;
#pragma unroll
        for (int j = 0; j < NTPW; ++j) {
          const int col = (wv * NTPW + j) * 16 + mr;
          // B 32x16 layout: lane = column, K {kb..kb+7, kb+16..kb+23} per lane.
          const v8h* pb = (const v8h*)(WL + (size_t)col * KS + kb);
          v8h blo = pb[0];
          v8h bhi = pb[2];
          v16h bm = __builtin_shufflevector(blo, bhi, 0, 1, 2, 3, 4, 5, 6, 7,
                                            8, 9, 10, 11, 12, 13, 14, 15);
          acc[j] = __builtin_amdgcn_wmma_f32_16x16x32_f16(
              false, a, false, bm, (short)0, acc[j], false, false);
        }
      }
      // C layout: VGPR r holds row (r + 8*half), col = n*16 + lane%16.
#pragma unroll
      for (int j = 0; j < NTPW; ++j) {
        const int col = (wv * NTPW + j) * 16 + mr;
#pragma unroll
        for (int r = 0; r < 8; ++r)
          gates[(r + 8 * half) * GS + col] = acc[j][r];
      }
      __syncthreads();

      // ================= elementwise: i,f,g,o -> c,h ========================
      float* cL = cSt + l * MT * CS;
      _Float16* hL = hA + l * MT * AS;
#pragma unroll
      for (int it = 0; it < (MT * HID) / NTHR; ++it) {   // 10 iterations
        const int idx = it * NTHR + tid;
        const int m  = idx / HID;
        const int jj = idx - m * HID;
        const float gi = gates[m * GS + jj];
        const float gf = gates[m * GS + HID + jj];
        const float gg = gates[m * GS + 2 * HID + jj];
        const float go = gates[m * GS + 3 * HID + jj];
        const float si = sigmoid_fast(gi);
        const float sf = sigmoid_fast(gf);
        const float tg = tanhf(gg);
        const float so = sigmoid_fast(go);
        const float cc = sf * cL[m * CS + jj] + si * tg;
        cL[m * CS + jj] = cc;
        const float hh = so * tanhf(cc);
        hL[m * AS + jj] = (_Float16)hh;
        if (l == 2) atomicAdd(&racc[m], wpL[jj] * hh);   // fused projection
      }
      __syncthreads();
    }

    // ---- projection finalize: out[b][t] = tanh(wp.h2 + bp) ----
    if (tid < MT) {
      out[(size_t)(b0row + tid) * TSZ + t] = tanhf(racc[tid] + wpL[HID]);
      racc[tid] = 0.f;
    }
    __syncthreads();
  }
}

extern "C" void kernel_launch(void* const* d_in, const int* in_sizes, int n_in,
                              void* d_out, int out_size, void* d_ws,
                              size_t ws_size, hipStream_t stream) {
  const float* z    = (const float*)d_in[0];
  const float* Wih0 = (const float*)d_in[1];   // [400][1]
  const float* Whh0 = (const float*)d_in[2];   // [400][100]
  const float* b0   = (const float*)d_in[3];
  const float* Wih1 = (const float*)d_in[4];
  const float* Whh1 = (const float*)d_in[5];
  const float* b1   = (const float*)d_in[6];
  const float* Wih2 = (const float*)d_in[7];
  const float* Whh2 = (const float*)d_in[8];
  const float* b2   = (const float*)d_in[9];
  const float* Wp   = (const float*)d_in[10];  // [1][100]
  const float* bp   = (const float*)d_in[11];
  float* out = (float*)d_out;

  _Float16* W = (_Float16*)d_ws;               // 3*400*264*2 = 633,600 B

  const int total = 3 * G4 * KS;
  const int pblocks = (total + 255) / 256;
  hipLaunchKernelGGL(prep_weights, dim3(pblocks), dim3(256), 0, stream,
                     Whh0, Wih1, Whh1, Wih2, Whh2, W);
  hipLaunchKernelGGL(lstm3_fused, dim3(BSZ / MT), dim3(NTHR), SMEM_BYTES,
                     stream, z, W, b0, b1, b2, Wih0, Wp, bp, out);
}
